// Attention_32787780338153
// MI455X (gfx1250) — matrix operand order repaired and checked by reading, not compiled, not run on hardware
//
#include <hip/hip_runtime.h>

#define SEQ   2048
#define DIM   4096
#define NH    32
#define NKV   8
#define HD    128
#define KVDIM (NKV * HD)   // 1024

static constexpr float ATT_SCALE = 0.08838834764831845f;  // 128^-0.5

typedef __attribute__((ext_vector_type(16))) __bf16 v16bf;
typedef __attribute__((ext_vector_type(8)))  float  v8f;
typedef __attribute__((ext_vector_type(4)))  unsigned int v4u;
typedef __attribute__((ext_vector_type(8)))  int v8i;
typedef __attribute__((ext_vector_type(4)))  int v4i;

#if defined(__has_builtin)
#if __has_builtin(__builtin_amdgcn_tensor_load_to_lds) && \
    __has_builtin(__builtin_amdgcn_s_wait_tensorcnt)
#define HAVE_TDM 1
#endif
#endif
#ifndef HAVE_TDM
#define HAVE_TDM 0
#endif

union FragU  { v16bf v; unsigned int u[8]; };
union FragU2 { v16bf v; uint4 q[2]; };

__device__ __forceinline__ unsigned short f2bf(float f) {
  unsigned int u = __float_as_uint(f);
  u += 0x7FFFu + ((u >> 16) & 1u);          // round-to-nearest-even
  return (unsigned short)(u >> 16);
}

__device__ __forceinline__ v8f wmma_bf16(v16bf a, v16bf b, v8f c) {
  return __builtin_amdgcn_wmma_f32_16x16x32_bf16(
      /*neg_a=*/false, a, /*neg_b=*/false, b,
      /*c_mod=*/(short)0, c, /*reuse_a=*/false, /*reuse_b=*/false);
}

// A-matrix fragment (ISA 16-bit A layout): lane l -> M = row0 + (l&15);
// VGPR i holds bf16 pair at K = col0 + (i<4 ? 2i : 16+2(i-4)) + 8*(l>=16).
template<int STRIDE>
__device__ __forceinline__ v16bf load_fragA(const unsigned short* base, int row0, int col0) {
  FragU f;
  const int lane = threadIdx.x & 31;
  const int ln = lane & 15, half = lane >> 4;
  const unsigned short* rp = base + (size_t)(row0 + ln) * STRIDE + col0 + half * 8;
  f.u[0] = *(const unsigned int*)(rp + 0);
  f.u[1] = *(const unsigned int*)(rp + 2);
  f.u[2] = *(const unsigned int*)(rp + 4);
  f.u[3] = *(const unsigned int*)(rp + 6);
  f.u[4] = *(const unsigned int*)(rp + 16);
  f.u[5] = *(const unsigned int*)(rp + 18);
  f.u[6] = *(const unsigned int*)(rp + 20);
  f.u[7] = *(const unsigned int*)(rp + 22);
  return f.v;
}

// B-matrix fragment (per ISA sparse-B tables): lane l -> N = row0 + (l&15);
// lanes 0-15 hold K = col0+{0..7,16..23}, lanes 16-31 hold K = col0+{8..15,24..31}
// (same K order as load_fragA).  Two 8-elem bf16 runs/lane -> two b128 LDS loads.
// STRIDE must be a multiple of 8 elements (16B) for alignment.
template<int STRIDE>
__device__ __forceinline__ v16bf load_fragB(const unsigned short* base, int row0, int col0) {
  FragU2 f;
  const int lane = threadIdx.x & 31;
  const int ln = lane & 15, half = lane >> 4;
  const unsigned short* rp = base + (size_t)(row0 + ln) * STRIDE + col0 + half * 8;
  f.q[0] = *(const uint4*)(rp);
  f.q[1] = *(const uint4*)(rp + 16);
  return f.v;
}

// ---------------------------------------------------------------------------
// Tiled GEMM: C(f32, MxN) = A(MxK) * B(f32, KxN), A fp32 or bf16.
// Block tile 128x128, BK=32, 256 threads = 8 waves (4m x 2n),
// each wave computes 32x64 via 2x4 v_wmma_f32_16x16x32_bf16.
// ---------------------------------------------------------------------------
template<bool ABF16>
__global__ __launch_bounds__(256) void gemm_wmma(
    const void* __restrict__ Ap, const float* __restrict__ B,
    float* __restrict__ C, int M, int N, int K) {
  __shared__ unsigned short As[128][40];   // [m][k] bf16, padded to 16B mult
  __shared__ unsigned short Bs[128][40];   // [n][k] bf16 (transposed), padded

  const int t   = threadIdx.x;
  const int m0  = blockIdx.y * 128, n0 = blockIdx.x * 128;
  const int wv_ = t >> 5, lane = t & 31;
  const int wm  = wv_ >> 1, wn = wv_ & 1;
  const int ln  = lane & 15, half = lane >> 4;

  v8f acc[2][4];
#pragma unroll
  for (int i = 0; i < 2; ++i)
#pragma unroll
    for (int j = 0; j < 4; ++j)
#pragma unroll
      for (int r = 0; r < 8; ++r) acc[i][j][r] = 0.f;

  const int arow = t >> 1, acb = (t & 1) * 16;   // A staging: 16 elems/thread
  const int brow = t >> 3, bnb = (t & 7) * 16;   // B staging: 16 elems/thread

  for (int k0 = 0; k0 < K; k0 += 32) {
    if constexpr (ABF16) {
      const unsigned short* src =
          (const unsigned short*)Ap + (size_t)(m0 + arow) * K + k0 + acb;
      __builtin_prefetch(src + 32, 0, 0);
#pragma unroll
      for (int i = 0; i < 16; ++i) As[arow][acb + i] = src[i];
    } else {
      const float* src = (const float*)Ap + (size_t)(m0 + arow) * K + k0 + acb;
      __builtin_prefetch(src + 32, 0, 0);
#pragma unroll
      for (int i = 0; i < 16; ++i) As[arow][acb + i] = f2bf(src[i]);
    }
    {
      const float* src = B + (size_t)(k0 + brow) * N + n0 + bnb;
      __builtin_prefetch(src + (size_t)32 * N, 0, 0);
#pragma unroll
      for (int i = 0; i < 16; ++i) Bs[bnb + i][brow] = f2bf(src[i]);
    }
    __syncthreads();

    v16bf a0 = load_fragA<40>(&As[0][0], wm * 32, 0);
    v16bf a1 = load_fragA<40>(&As[0][0], wm * 32 + 16, 0);
#pragma unroll
    for (int j = 0; j < 4; ++j) {
      v16bf bj = load_fragB<40>(&Bs[0][0], wn * 64 + j * 16, 0);
      acc[0][j] = wmma_bf16(a0, bj, acc[0][j]);
      acc[1][j] = wmma_bf16(a1, bj, acc[1][j]);
    }
    __syncthreads();
  }

#pragma unroll
  for (int i = 0; i < 2; ++i)
#pragma unroll
    for (int j = 0; j < 4; ++j)
#pragma unroll
      for (int r = 0; r < 8; ++r) {
        const int grow = m0 + wm * 32 + i * 16 + r + half * 8;
        const int gcol = n0 + wn * 64 + j * 16 + ln;
        C[(size_t)grow * N + gcol] = acc[i][j][r];
      }
}

// ---------------------------------------------------------------------------
// RoPE (interleaved pairs) + fp32 -> bf16 convert.
// ---------------------------------------------------------------------------
__global__ __launch_bounds__(256) void rope_cvt(
    const float* __restrict__ tin, const float* __restrict__ fc,
    const float* __restrict__ fs, unsigned short* __restrict__ tout, int H) {
  size_t i = (size_t)blockIdx.x * blockDim.x + threadIdx.x;
  if (i >= (size_t)SEQ * H * (HD / 2)) return;
  const int d2 = (int)(i & 63);
  size_t sh = i >> 6;
  const int hh = (int)(sh % H);
  const size_t s = sh / H;
  const float c = fc[s * (HD / 2) + d2];
  const float sn = fs[s * (HD / 2) + d2];
  const size_t base = ((size_t)s * H + hh) * HD + 2 * d2;
  const float t1 = tin[base], t2 = tin[base + 1];
  tout[base]     = f2bf(t1 * c - t2 * sn);
  tout[base + 1] = f2bf(t1 * sn + t2 * c);
}

__global__ __launch_bounds__(256) void cvt_bf16(
    const float* __restrict__ in, unsigned short* __restrict__ out, size_t n) {
  size_t i = (size_t)blockIdx.x * blockDim.x + threadIdx.x;
  if (i < n) out[i] = f2bf(in[i]);
}

// ---------------------------------------------------------------------------
// Flash attention, WMMA bf16.  Grid (SEQ/64, NH); 128 threads = 4 waves,
// each wave owns 16 query rows; 32 keys/iteration.
// K tile staged by the Tensor Data Mover (2D D#, LDS padding = row stride
// 136 u16), drained with s_wait_tensorcnt; V staged transposed by VALU.
// ---------------------------------------------------------------------------
__global__ __launch_bounds__(128) void attn_wmma(
    const unsigned short* __restrict__ Q, const unsigned short* __restrict__ Kb,
    const unsigned short* __restrict__ V, unsigned short* __restrict__ O) {
  __shared__ unsigned short Ks[32][136];    // [key][d]  (136 = 128 + 4 DWORD pad)
  __shared__ unsigned short Vt[128][40];    // [d][key]  (transposed)
  __shared__ unsigned short Ps[4][16][40];  // per-wave P tile 16x32

  const int t = threadIdx.x;
  const int h = blockIdx.y;
  const int hk = h >> 2;                    // GQA: 4 query heads per kv head
  const int w = t >> 5, lane = t & 31, ln = lane & 15, half = lane >> 4;
  const int qbase = blockIdx.x * 64 + w * 16;

  // Preload Q fragments straight from global (row stride NH*HD = 4096).
  v16bf qf[4];
#pragma unroll
  for (int c = 0; c < 4; ++c)
    qf[c] = load_fragA<NH * HD>(Q + (size_t)h * HD, qbase, c * 32);

  v8f acc[8];
  float mrow[8], lrow[8];
#pragma unroll
  for (int r = 0; r < 8; ++r) {
    mrow[r] = -1e30f;
    lrow[r] = 0.f;
#pragma unroll
    for (int c = 0; c < 8; ++c) acc[c][r] = 0.f;
  }

#if HAVE_TDM
  // D# group 1 (static): data_size=2B, pad_enable, pad_interval=64 DWORDs,
  // pad_amount=4 DWORDs, tensor_dim0=128, tensor_dim1=32, tile=128x32,
  // dim0 stride = NKV*HD = 1024 elements.
  v8i g1;
  g1[0] = (int)((1u << 16) | (1u << 20) | (5u << 22) | (3u << 25));
  g1[1] = (int)(128u << 16);
  g1[2] = (int)(32u << 16);
  g1[3] = (int)(128u << 16);
  g1[4] = 32;
  g1[5] = 1024;
  g1[6] = 0;
  g1[7] = 0;
  v4i gz4; gz4[0] = 0; gz4[1] = 0; gz4[2] = 0; gz4[3] = 0;
  v8i gz8;
#pragma unroll
  for (int i = 0; i < 8; ++i) gz8[i] = 0;
  const unsigned ldsKs = (unsigned)(uintptr_t)&Ks[0][0];
#endif

  const int skk = t >> 2, sdb = (t & 3) * 32;   // V staging coords

  for (int jt = 0; jt < SEQ / 32; ++jt) {
#if HAVE_TDM
    if (t < 32) {  // wave 0 drives the TDM; EXEC is ignored by tensor ops
      unsigned long long ga = (unsigned long long)(uintptr_t)Kb +
          (((unsigned long long)(jt * 32) * NKV + hk) * HD) * 2ull;
      v4u g0;
      g0[0] = 1u;                       // count = 1 descriptor
      g0[1] = ldsKs;                    // lds_addr
      g0[2] = (unsigned)ga;             // global_addr[31:0]
      g0[3] = (unsigned)(ga >> 32) | (2u << 30);  // addr[56:32] | type=2
      __builtin_amdgcn_tensor_load_to_lds(g0, g1, gz4, gz4, gz8, 0);
      __builtin_amdgcn_s_wait_tensorcnt(0);
    }
#else
    {  // fallback: manual K tile stage [32 keys][128 d]
      const unsigned int* src = (const unsigned int*)(
          Kb + ((size_t)(jt * 32 + skk) * NKV + hk) * HD + sdb);
      unsigned int* dst = (unsigned int*)&Ks[skk][sdb];
#pragma unroll
      for (int i = 0; i < 16; ++i) dst[i] = src[i];
    }
#endif
    {  // stage V tile transposed [128 d][32 keys]
      const unsigned short* src =
          V + ((size_t)(jt * 32 + skk) * NKV + hk) * HD + sdb;
#pragma unroll
      for (int i = 0; i < 32; ++i) Vt[sdb + i][skk] = src[i];
    }
    __syncthreads();

    // S = Q * K^T for 16 rows x 32 keys (two 16x16 C tiles)
    v8f s0, s1;
#pragma unroll
    for (int r = 0; r < 8; ++r) { s0[r] = 0.f; s1[r] = 0.f; }
#pragma unroll
    for (int c = 0; c < 4; ++c) {
      v16bf kf0 = load_fragB<136>(&Ks[0][0], 0, c * 32);
      v16bf kf1 = load_fragB<136>(&Ks[0][0], 16, c * 32);
      s0 = wmma_bf16(qf[c], kf0, s0);
      s1 = wmma_bf16(qf[c], kf1, s1);
    }

    // Online softmax.  C-layout: reg r, half g -> row M = r + 8g, col = ln.
    unsigned short* pw = &Ps[w][0][0];
#pragma unroll
    for (int r = 0; r < 8; ++r) {
      float v0 = s0[r] * ATT_SCALE, v1 = s1[r] * ATT_SCALE;
      float rm = fmaxf(v0, v1);
#pragma unroll
      for (int off = 1; off < 16; off <<= 1)
        rm = fmaxf(rm, __shfl_xor(rm, off, 16));
      const float mn = fmaxf(mrow[r], rm);
      const float cr = __expf(mrow[r] - mn);
      const float p0 = __expf(v0 - mn), p1 = __expf(v1 - mn);
      float rs = p0 + p1;
#pragma unroll
      for (int off = 1; off < 16; off <<= 1) rs += __shfl_xor(rs, off, 16);
      lrow[r] = lrow[r] * cr + rs;
      mrow[r] = mn;
#pragma unroll
      for (int c = 0; c < 8; ++c) acc[c][r] *= cr;
      pw[(r + half * 8) * 40 + ln]      = f2bf(p0);
      pw[(r + half * 8) * 40 + 16 + ln] = f2bf(p1);
    }
    // Same-wave LDS RAW: drain DS counter before re-reading P in A-layout.
    asm volatile("s_wait_dscnt 0x0" ::: "memory");

    v16bf pf = load_fragA<40>(pw, 0, 0);
#pragma unroll
    for (int c = 0; c < 8; ++c) {
      v16bf vf = load_fragB<40>(&Vt[0][0], c * 16, 0);
      acc[c] = wmma_bf16(pf, vf, acc[c]);
    }
    __syncthreads();
  }

  // Normalize and store bf16 output, layout (s, h*128 + d).
#pragma unroll
  for (int r = 0; r < 8; ++r) {
    const float inv = 1.f / lrow[r];
    const int grow = qbase + r + half * 8;
    unsigned short* orow = O + (size_t)grow * (NH * HD) + h * HD;
#pragma unroll
    for (int c = 0; c < 8; ++c) orow[c * 16 + ln] = f2bf(acc[c][r] * inv);
  }
}

// ---------------------------------------------------------------------------
extern "C" void kernel_launch(void* const* d_in, const int* in_sizes, int n_in,
                              void* d_out, int out_size, void* d_ws, size_t ws_size,
                              hipStream_t stream) {
  const float* x  = (const float*)d_in[0];
  const float* fc = (const float*)d_in[1];
  const float* fs = (const float*)d_in[2];
  const float* wq = (const float*)d_in[3];
  const float* wk = (const float*)d_in[4];
  const float* wv = (const float*)d_in[5];
  const float* wo = (const float*)d_in[6];
  float* out = (float*)d_out;

  char* ws = (char*)d_ws;
  float*          qf = (float*)(ws);                              // 32 MB  q fp32
  float*          kf = (float*)(ws + (size_t)33554432);           //  8 MB  k fp32
  float*          vf = (float*)(ws + (size_t)41943040);           //  8 MB  v fp32
  unsigned short* qb = (unsigned short*)(ws + (size_t)50331648);  // 16 MB  q bf16
  unsigned short* kb = (unsigned short*)(ws + (size_t)67108864);  //  4 MB  k bf16
  unsigned short* vb = (unsigned short*)(ws + (size_t)71303168);  //  4 MB  v bf16
  unsigned short* ab = (unsigned short*)(ws + (size_t)75497472);  // 16 MB  attn out bf16

  // QKV projections (fp32 A/B -> bf16 WMMA -> fp32 C)
  gemm_wmma<false><<<dim3(DIM / 128,   SEQ / 128), 256, 0, stream>>>(x, wq, qf, SEQ, DIM,   DIM);
  gemm_wmma<false><<<dim3(KVDIM / 128, SEQ / 128), 256, 0, stream>>>(x, wk, kf, SEQ, KVDIM, DIM);
  gemm_wmma<false><<<dim3(KVDIM / 128, SEQ / 128), 256, 0, stream>>>(x, wv, vf, SEQ, KVDIM, DIM);

  // RoPE + bf16 conversion
  rope_cvt<<<(SEQ * NH  * (HD / 2)) / 256, 256, 0, stream>>>(qf, fc, fs, qb, NH);
  rope_cvt<<<(SEQ * NKV * (HD / 2)) / 256, 256, 0, stream>>>(kf, fc, fs, kb, NKV);
  cvt_bf16<<<(SEQ * KVDIM) / 256, 256, 0, stream>>>(vf, vb, (size_t)SEQ * KVDIM);

  // Flash attention
  attn_wmma<<<dim3(SEQ / 64, NH), 128, 0, stream>>>(qb, kb, vb, ab);

  // Output projection (bf16 A from attention, fp32 weights)
  gemm_wmma<true><<<dim3(DIM / 128, SEQ / 128), 256, 0, stream>>>(ab, wo, out, SEQ, DIM, DIM);
}
